// AttentionModel_83288005804250
// MI455X (gfx1250) — matrix operand order repaired and checked by reading, not compiled, not run on hardware
//
#include <hip/hip_runtime.h>

typedef __bf16 bf16_t;
typedef __attribute__((ext_vector_type(16))) __bf16 v16bf;
typedef __attribute__((ext_vector_type(8)))  __bf16 v8bf;
typedef __attribute__((ext_vector_type(4)))  __bf16 v4bf;
typedef __attribute__((ext_vector_type(8)))  float   v8f;
typedef __attribute__((ext_vector_type(4)))  float   v4f;

#define DD 1024
#define RR 64
#define BB 8
#define SS 4096
#define LOG2E 1.44269504088896340736f

#define KT 32
#define DCHUNK 256
#define VT_STRIDE 80       /* 32 keys * 2B + 16B pad (TDM: pad_interval=3, pad_amount=3) */
#define VT_BYTES (DD * VT_STRIDE)
#define PB_STRIDE 80
#define QT_STRIDE 2064

#if defined(__has_builtin)
#if __has_builtin(__builtin_amdgcn_tensor_load_to_lds)
#define HAS_TDM 1
#endif
#endif
#ifndef HAS_TDM
#define HAS_TDM 0
#warning "gfx1250 TDM builtin not available: falling back to cooperative LDS copy"
#endif

static __device__ __forceinline__ v16bf bfcat(v8bf lo, v8bf hi) {
  union { v16bf w; v8bf h[2]; } u;
  u.h[0] = lo; u.h[1] = hi;
  return u.w;
}

static __device__ __forceinline__ v8f wmma_bf16(v16bf a, v16bf b, v8f c) {
  // D = A(16x32 bf16) * B(32x16 bf16) + C(16x16 f32)
  return __builtin_amdgcn_wmma_f32_16x16x32_bf16(false, a, false, b, (short)0, c, false, false);
}

#if HAS_TDM
typedef unsigned int v4u_t __attribute__((ext_vector_type(4)));
typedef int          v8i_t __attribute__((ext_vector_type(8)));
typedef int          v4i_t __attribute__((ext_vector_type(4)));

// Issue one TDM load: 2D tile (tile_dim0=32 keys contiguous, tile_dim1=1024 d rows)
// out of tensor xvT[b] (tensor_dim0=4096 keys, row stride 4096), into LDS at lds_off,
// with 16B padding inserted every 64B -> LDS row stride = VT_STRIDE.
// This toolchain exposes the 6-arg builtin:
//   (uint32x4 g0, int32x8 g1, int32x4 g2, int32x4 g3, int32x8 extra, i32 cpol)
static __device__ __forceinline__ void tdm_issue_vt(unsigned lds_off, const bf16_t* gsrc) {
  const unsigned long long ga = (unsigned long long)(uintptr_t)gsrc;
  v4u_t g0;
  g0[0] = 1u;                                                // count=1 (valid descriptor)
  g0[1] = lds_off;                                           // lds_addr (bytes)
  g0[2] = (unsigned)(ga & 0xFFFFFFFFu);                      // global_addr[31:0]
  g0[3] = (unsigned)((ga >> 32) & 0x01FFFFFFu) | (2u << 30); // global_addr[56:32] | type=2
  v8i_t g1;
  g1[0] = (int)0x06D10000;        // data_size=1(2B), pad_enable=1, pad_interval=3, pad_amount=3
  g1[1] = (int)(0x1000u << 16);   // tensor_dim0[15:0] = 4096 at bits[63:48]
  g1[2] = (int)(0x0400u << 16);   // tensor_dim0 hi=0 | tensor_dim1[15:0] = 1024
  g1[3] = (int)(32u << 16);       // tensor_dim1 hi=0 | tile_dim0 = 32
  g1[4] = (int)1024;              // tile_dim1 = 1024, tile_dim2 = 0
  g1[5] = (int)4096;              // tensor_dim0_stride[31:0] = 4096
  g1[6] = 0;                      // stride hi, tensor_dim1_stride lo
  g1[7] = 0;                      // tensor_dim1_stride hi
  v4i_t gz; gz[0] = 0; gz[1] = 0; gz[2] = 0; gz[3] = 0;
  v8i_t gz8; gz8[0] = 0; gz8[1] = 0; gz8[2] = 0; gz8[3] = 0;
  gz8[4] = 0; gz8[5] = 0; gz8[6] = 0; gz8[7] = 0;
  __builtin_amdgcn_tensor_load_to_lds(g0, g1, gz, gz, gz8, 0);
}
#endif

// ---------------------------------------------------------------------------
// Phase 1b: x [B][S][D] f32 -> xvT [B][D][S] bf16 (tiled transpose via LDS)
// ---------------------------------------------------------------------------
__global__ __launch_bounds__(256) void cvtT_kernel(const float* __restrict__ x,
                                                   bf16_t* __restrict__ xvT) {
  __shared__ bf16_t tile[32][40];  // [d][key], padded
  const int b = blockIdx.z;
  const int key0 = blockIdx.x * 32;
  const int d0 = blockIdx.y * 32;
  const int t = threadIdx.x;
  {
    const int key = t >> 3;        // 0..31
    const int dq = (t & 7) * 4;    // 0,4,...,28
    const v4f f = *(const v4f*)(x + ((long)b * SS + key0 + key) * DD + d0 + dq);
#pragma unroll
    for (int j = 0; j < 4; j++) tile[dq + j][key] = (bf16_t)f[j];
  }
  __syncthreads();
  {
    const int d = t >> 3;          // 0..31
    const int kq = (t & 7) * 4;    // 0..28
    v4bf o;
#pragma unroll
    for (int j = 0; j < 4; j++) o[j] = tile[d][kq + j];
    *(v4bf*)(xvT + ((long)b * DD + d0 + d) * SS + key0 + kq) = o;
  }
}

// ---------------------------------------------------------------------------
// Phase 1a: xq = x @ (Q / sqrt(D)) in bf16, via WMMA. One 16x64 tile per wave.
// ---------------------------------------------------------------------------
__global__ __launch_bounds__(256) void proj_kernel(const float* __restrict__ x,
                                                   const float* __restrict__ Qm,
                                                   bf16_t* __restrict__ xqo) {
  extern __shared__ char smem[];
  const int t = threadIdx.x;

  // Stage Q^T (scaled) into LDS: Qt[r][feat], row stride QT_STRIDE bytes
  for (int idx = t; idx < DD * RR; idx += 256) {
    const int feat = idx >> 6;
    const int r = idx & 63;
    const float qv = Qm[idx] * 0.03125f;  // 1/sqrt(1024)
    *(bf16_t*)(smem + (long)r * QT_STRIDE + feat * 2) = (bf16_t)qv;
  }
  __syncthreads();

  const int wid = t >> 5;
  const int lane = t & 31;
  const int half = lane >> 4;
  const int ln = lane & 15;
  const long gtile = (long)blockIdx.x * 8 + wid;
  const float* xrow = x + (gtile * 16 + ln) * DD;

  v8f zero = {};
  v8f acc[4] = {zero, zero, zero, zero};

#pragma unroll 1
  for (int k = 0; k < DD; k += 32) {
    v8bf alo, ahi;
    {
      const v4f f0 = *(const v4f*)(xrow + k + 8 * half);
      const v4f f1 = *(const v4f*)(xrow + k + 8 * half + 4);
      const v4f f2 = *(const v4f*)(xrow + k + 16 + 8 * half);
      const v4f f3 = *(const v4f*)(xrow + k + 16 + 8 * half + 4);
#pragma unroll
      for (int j = 0; j < 4; j++) {
        alo[j]     = (bf16_t)f0[j];
        alo[j + 4] = (bf16_t)f1[j];
        ahi[j]     = (bf16_t)f2[j];
        ahi[j + 4] = (bf16_t)f3[j];
      }
    }
    const v16bf a = bfcat(alo, ahi);
#pragma unroll
    for (int rt = 0; rt < 4; rt++) {
      const char* qrow = smem + (long)(rt * 16 + ln) * QT_STRIDE + (k + 8 * half) * 2;
      const v16bf bq = bfcat(*(const v8bf*)qrow, *(const v8bf*)(qrow + 32));
      acc[rt] = wmma_bf16(a, bq, acc[rt]);
    }
  }

#pragma unroll
  for (int rt = 0; rt < 4; rt++)
#pragma unroll
    for (int r = 0; r < 8; r++)
      xqo[(gtile * 16 + r + 8 * half) * RR + rt * 16 + ln] = (bf16_t)acc[rt][r];
}

// ---------------------------------------------------------------------------
// Phase 2: fused flash attention, TDM double-buffered V^T staging.
// Block = 8 waves = (2 q-subtiles) x (4 d-chunks of 256).
// ---------------------------------------------------------------------------
__global__ __launch_bounds__(256) void attn_kernel(const bf16_t* __restrict__ xq,
                                                   const bf16_t* __restrict__ xvT,
                                                   float* __restrict__ y) {
  extern __shared__ char smem[];
  char* PB = smem + 2 * VT_BYTES;       // P buffers: 8 waves x 16 q-rows x PB_STRIDE

  const int t = threadIdx.x;
  const int wid = t >> 5;
  const int lane = t & 31;
  const int half = lane >> 4;
  const int ln = lane & 15;
  const int b = blockIdx.y;
  const int qsub = wid & 1;
  const int dbase = (wid >> 1) * DCHUNK;
  const long qrow0 = (long)b * SS + (long)blockIdx.x * 32 + qsub * 16;
  char* Pw = PB + (long)wid * 16 * PB_STRIDE;

  // Preload B-fragments of q: B[feat][q], lane -> q col (ln), elems -> feat
  const bf16_t* xqrow = xq + (qrow0 + ln) * RR;
  const v16bf bq0 = bfcat(*(const v8bf*)(xqrow + 8 * half),
                          *(const v8bf*)(xqrow + 16 + 8 * half));
  const v16bf bq1 = bfcat(*(const v8bf*)(xqrow + 32 + 8 * half),
                          *(const v8bf*)(xqrow + 48 + 8 * half));

  v8f zero = {};
  v8f acc[16];
#pragma unroll
  for (int i = 0; i < 16; i++) acc[i] = zero;

  float m_old = -__builtin_inff();
  float l_run = 0.0f;

  const bf16_t* xvTb = xvT + (long)b * DD * SS;  // [D][S]
  const bf16_t* xqb = xq + (long)b * SS * RR;

#if HAS_TDM
  // prologue: issue tile 0 into buffer 0
  if (wid == 0) tdm_issue_vt((unsigned)(uintptr_t)(void*)smem, xvTb);
#endif

#pragma unroll 1
  for (int it = 0; it < SS / KT; it++) {
    const int key0 = it * KT;
    char* VT = smem + (unsigned)(it & 1) * VT_BYTES;

#if HAS_TDM
    if (it + 1 < SS / KT) {
      if (wid == 0)
        tdm_issue_vt((unsigned)(uintptr_t)(void*)(smem + (unsigned)((it + 1) & 1) * VT_BYTES),
                     xvTb + key0 + KT);
      __builtin_amdgcn_s_wait_tensorcnt(1);  // tile `it` complete, `it+1` may be in flight
    } else {
      __builtin_amdgcn_s_wait_tensorcnt(0);
    }
    __syncthreads();  // all waves observe VT[it&1] ready
#else
    __syncthreads();
    // fallback cooperative copy: 1024 rows x 64B, padded to VT_STRIDE
#pragma unroll 4
    for (int rr = 0; rr < 16; rr++) {
      const int d = (t >> 2) + rr * 64;
      const int kc = (t & 3) * 8;
      *(v8bf*)(VT + (long)d * VT_STRIDE + kc * 2) =
          *(const v8bf*)(xvTb + (long)d * SS + key0 + kc);
    }
    __syncthreads();
#endif

    // ---- scores, transposed: S^T(32 keys x 16 q) = XQ_keys @ XQ_q^T ----
    v8f st[2];
#pragma unroll
    for (int mt = 0; mt < 2; mt++) {
      const bf16_t* krow = xqb + (long)(key0 + mt * 16 + ln) * RR;
      const v16bf a0 = bfcat(*(const v8bf*)(krow + 8 * half),
                             *(const v8bf*)(krow + 16 + 8 * half));
      const v16bf a1 = bfcat(*(const v8bf*)(krow + 32 + 8 * half),
                             *(const v8bf*)(krow + 48 + 8 * half));
      v8f c = zero;
      c = wmma_bf16(a0, bq0, c);
      c = wmma_bf16(a1, bq1, c);
      st[mt] = c;  // rows = keys mt*16 + r + 8*half, col = q (ln)
    }

    // ---- online softmax (lane owns q col ln) ----
    float tmax = -__builtin_inff();
#pragma unroll
    for (int mt = 0; mt < 2; mt++)
#pragma unroll
      for (int r = 0; r < 8; r++) tmax = fmaxf(tmax, st[mt][r]);
    tmax = fmaxf(tmax, __shfl_xor(tmax, 16, 32));
    const float m_new = fmaxf(m_old, tmax);
    const float corr = exp2f((m_old - m_new) * LOG2E);

    float tsum = 0.0f;
#pragma unroll
    for (int mt = 0; mt < 2; mt++) {
#pragma unroll
      for (int r = 0; r < 8; r += 2) {
        const float p0 = exp2f((st[mt][r]     - m_new) * LOG2E);
        const float p1 = exp2f((st[mt][r + 1] - m_new) * LOG2E);
        tsum += p0 + p1;
        union { unsigned u; unsigned short s[2]; } pk;
        union { bf16_t bv; unsigned short s; } c0, c1;
        c0.bv = (bf16_t)p0; c1.bv = (bf16_t)p1;
        pk.s[0] = c0.s; pk.s[1] = c1.s;
        const int key = mt * 16 + 8 * half + r;
        *(unsigned*)(Pw + (long)ln * PB_STRIDE + key * 2) = pk.u;  // P[q=ln][key]
      }
    }
    tsum += __shfl_xor(tsum, 16, 32);
    l_run = l_run * corr + tsum;
    m_old = m_new;

    // lane-transposed LDS handoff within the wave
    asm volatile("s_wait_dscnt 0x0" ::: "memory");

    // ---- rescale accumulators (row = q = r + 8*half) ----
#pragma unroll
    for (int r = 0; r < 8; r++) {
      const float cr = __shfl(corr, r + 8 * half, 16);
#pragma unroll
      for (int nt = 0; nt < 16; nt++) acc[nt][r] *= cr;
    }

    // ---- P A-fragment: lane -> q row (ln); elems -> keys ----
    const char* prow = Pw + (long)ln * PB_STRIDE;
    const v16bf pa = bfcat(*(const v8bf*)(prow + 16 * half),
                           *(const v8bf*)(prow + 32 + 16 * half));

    // ---- P @ V: 16 n-tiles of 16 d-cols each ----
#pragma unroll
    for (int nt = 0; nt < 16; nt++) {
      const char* vrow = VT + (long)(dbase + nt * 16 + ln) * VT_STRIDE + 16 * half;
      const v16bf vb = bfcat(*(const v8bf*)vrow, *(const v8bf*)(vrow + 32));
      acc[nt] = wmma_bf16(pa, vb, acc[nt]);
    }

    __syncthreads();  // done reading VT[it&1]; next issue may overwrite this buffer
  }

  // ---- epilogue: y = acc / l ----
  const float linv = 1.0f / l_run;
#pragma unroll
  for (int r = 0; r < 8; r++) {
    const float li = __shfl(linv, r + 8 * half, 16);
    float* yr = y + (qrow0 + r + 8 * half) * DD + dbase + ln;
#pragma unroll
    for (int nt = 0; nt < 16; nt++) yr[nt * 16] = acc[nt][r] * li;
  }
}

// ---------------------------------------------------------------------------
extern "C" void kernel_launch(void* const* d_in, const int* in_sizes, int n_in,
                              void* d_out, int out_size, void* d_ws, size_t ws_size,
                              hipStream_t stream) {
  (void)in_sizes; (void)n_in; (void)out_size; (void)ws_size;
  const float* x  = (const float*)d_in[0];
  const float* Qm = (const float*)d_in[1];
  float* y = (float*)d_out;

  bf16_t* xq  = (bf16_t*)d_ws;                                       // 4 MB
  bf16_t* xvT = (bf16_t*)((char*)d_ws + (size_t)BB * SS * RR * 2);   // 67 MB, [B][D][S]

  cvtT_kernel<<<dim3(SS / 32, DD / 32, BB), 256, 0, stream>>>(x, xvT);
  proj_kernel<<<(BB * SS / 16) / 8, 256, RR * QT_STRIDE, stream>>>(x, Qm, xq);
  attn_kernel<<<dim3(SS / 32, BB), 256, 2 * VT_BYTES + 8 * 16 * PB_STRIDE, stream>>>(xq, xvT, y);
}
